// MoEQFunction_23089744183763
// MI455X (gfx1250) — matrix operand order repaired and checked by reading, not compiled, not run on hardware
//
#include <hip/hip_runtime.h>

// ---------------------------------------------------------------------------
// MoE Q-function forward, bf16 WMMA (gfx1250).
// All heavy GEMMs -> v_wmma_f32_16x16x32_bf16. D padded 400 -> 416 (13 K-steps).
// Weights stored transposed [N][Ks] bf16 with zeroed K-pads.
// Wave tile 32x64 (2x4 WMMA accs), block = 8 waves stacked in M (256x64).
// K loop software-pipelined with 2 ping-pong fragment sets (no rotation copies).
// ---------------------------------------------------------------------------

typedef __bf16 bf16;
typedef __attribute__((ext_vector_type(8)))  bf16  bf16x8;
typedef __attribute__((ext_vector_type(16))) bf16  bf16x16;
typedef __attribute__((ext_vector_type(8)))  float floatx8;

#define BATCH   16384
#define DPAD    416
#define NV      400
#define NEXP    8
#define BM      256
#define BN      64

// ---- fragment loaders ------------------------------------------------------
// A (16x32 bf16): lane m = l&15, khalf = l>>4; regs cover K = {0..7,16..23}+8*khalf
__device__ inline bf16x16 frag_a(const bf16* p) {
  bf16x8 lo = *(const bf16x8*)(p);
  bf16x8 hi = *(const bf16x8*)(p + 16);
  bf16x16 f;
#pragma unroll
  for (int i = 0; i < 8; ++i) { f[i] = lo[i]; f[i + 8] = hi[i]; }
  return f;
}
// B (32x16 bf16): lane n = l&15, khalf = l>>4; regs cover K = {0..7,16..23}+8*khalf
__device__ inline bf16x16 frag_b(const bf16* p) {
  bf16x8 lo = *(const bf16x8*)(p);
  bf16x8 hi = *(const bf16x8*)(p + 16);
  bf16x16 f;
#pragma unroll
  for (int i = 0; i < 8; ++i) { f[i] = lo[i]; f[i + 8] = hi[i]; }
  return f;
}

struct Frags { bf16x16 a0, a1, b[4]; };

__device__ inline void load_frags(Frags& f, const bf16* arow0, const bf16* arow1,
                                  const bf16* const (&brow)[4], int ka, int kb) {
  f.a0 = frag_a(arow0 + ka);
  f.a1 = frag_a(arow1 + ka);
#pragma unroll
  for (int nt = 0; nt < 4; ++nt) f.b[nt] = frag_b(brow[nt] + kb);
}

__device__ inline void wmma_step(floatx8 (&acc)[2][4], const Frags& f) {
#pragma unroll
  for (int nt = 0; nt < 4; ++nt) {
    acc[0][nt] = __builtin_amdgcn_wmma_f32_16x16x32_bf16(false, f.a0, false, f.b[nt], (short)0, acc[0][nt], false, false);
    acc[1][nt] = __builtin_amdgcn_wmma_f32_16x16x32_bf16(false, f.a1, false, f.b[nt], (short)0, acc[1][nt], false, false);
  }
}

// ---- generic batched GEMM:  C[z] = act( A[z][M,Ks] @ Wt[z][N,Ks]^T + bias ) -
__global__ __launch_bounds__(256) void gemm_bf16_wmma(
    const bf16* __restrict__ A, const bf16* __restrict__ Wt,
    const float* __restrict__ bias, bf16* __restrict__ C,
    int Nvalid, int Ks, int lda, int ldc,
    long aBatch, long wBatch, long biasBatch, long cBatch, int relu)
{
  const int z = blockIdx.z;
  A  += (size_t)z * aBatch;
  Wt += (size_t)z * wBatch;
  if (bias) bias += (size_t)z * biasBatch;
  C  += (size_t)z * cBatch;

  const int lane  = threadIdx.x & 31;
  const int wave  = threadIdx.x >> 5;                    // 8 waves stacked in M
  const int m0 = blockIdx.x * BM + wave * 32;
  const int n0 = blockIdx.y * BN;
  const int l15   = lane & 15;
  const int khalf = lane >> 4;

  floatx8 acc[2][4];
#pragma unroll
  for (int mt = 0; mt < 2; ++mt)
#pragma unroll
    for (int nt = 0; nt < 4; ++nt) acc[mt][nt] = (floatx8){};

  const bf16* arow0 = A + (size_t)(m0 + l15) * lda;
  const bf16* arow1 = arow0 + (size_t)16 * lda;
  int  nc[4];
  bool bv[4];
  const bf16* brow[4];
#pragma unroll
  for (int nt = 0; nt < 4; ++nt) {
    nc[nt] = n0 + nt * 16 + l15;
    bv[nt] = nc[nt] < Nvalid;
    brow[nt] = Wt + (size_t)(bv[nt] ? nc[nt] : 0) * Ks;  // clamp: always valid mem
  }

  const int nK = Ks >> 5;                                // >= 2 for all layers
  const int kaOff = 8 * khalf;
  const int kbOff = 8 * khalf;

  // ping-pong pipeline: load f1 while computing f0, and vice versa
  Frags f0, f1;
  load_frags(f0, arow0, arow1, brow, kaOff, kbOff);      // kt = 0
  int kt = 0;
  for (; kt + 2 < nK; kt += 2) {
    load_frags(f1, arow0, arow1, brow, (kt + 1) * 32 + kaOff, (kt + 1) * 32 + kbOff);
    wmma_step(acc, f0);
    load_frags(f0, arow0, arow1, brow, (kt + 2) * 32 + kaOff, (kt + 2) * 32 + kbOff);
    wmma_step(acc, f1);
  }
  if (kt + 1 < nK) {                                     // two steps remain
    load_frags(f1, arow0, arow1, brow, (kt + 1) * 32 + kaOff, (kt + 1) * 32 + kbOff);
    wmma_step(acc, f0);
    wmma_step(acc, f1);
  } else {                                               // one step remains
    wmma_step(acc, f0);
  }

  // epilogue: C/D layout -> lane col n = nc[nt]; reg r -> row m0+mt*16+r+8*khalf
#pragma unroll
  for (int nt = 0; nt < 4; ++nt) {
    if (!bv[nt]) continue;
    const float bvx = bias ? bias[nc[nt]] : 0.0f;
#pragma unroll
    for (int r = 0; r < 8; ++r) {
      const int mA = m0 + r + 8 * khalf;
      const int mB = mA + 16;
      float v0 = acc[0][nt][r] + bvx; if (relu) v0 = fmaxf(v0, 0.0f);
      float v1 = acc[1][nt][r] + bvx; if (relu) v1 = fmaxf(v1, 0.0f);
      C[(size_t)mA * ldc + nc[nt]] = (bf16)v0;
      C[(size_t)mB * ldc + nc[nt]] = (bf16)v1;
    }
  }
}

// ---- input concat + bf16 conversion: XA[B,64] = [obs|act|0pad] --------------
__global__ void build_x(const float* __restrict__ obs, const float* __restrict__ act,
                        bf16* __restrict__ XA, int B)
{
  int idx = blockIdx.x * blockDim.x + threadIdx.x;
  if (idx >= B * 64) return;
  int b = idx >> 6, c = idx & 63;
  float v = 0.0f;
  if (c < 39)      v = obs[b * 39 + c];
  else if (c < 43) v = act[b * 4 + (c - 39)];
  XA[idx] = (bf16)v;
}

// ---- weight convert+transpose to [N][Ks] bf16 with zero K-pads --------------
// srcNK=1: source laid out [N][Kdim] (key/value mats). srcNK=0: [Kdim][N].
__global__ void convert_wt(const float* __restrict__ W, bf16* __restrict__ Wt,
                           int Kdim, int N, int Ks, int srcNK,
                           long srcBatch, long dstBatch, int total)
{
  int idx = blockIdx.x * blockDim.x + threadIdx.x;
  if (idx >= total) return;
  int k = idx % Ks;
  int t = idx / Ks;
  int n = t % N;
  int z = t / N;
  float v = 0.0f;
  if (k < Kdim)
    v = srcNK ? W[(size_t)z * srcBatch + (size_t)n * Kdim + k]
              : W[(size_t)z * srcBatch + (size_t)k * N + n];
  Wt[(size_t)z * dstBatch + (size_t)n * Ks + k] = (bf16)v;
}

// ---- scores -> softmax -> weighted sum (one wave32 per row) -----------------
__global__ __launch_bounds__(256) void attn_combine(
    const bf16* __restrict__ KEY, const bf16* __restrict__ VAL,
    const int* __restrict__ task, const float* __restrict__ tq,
    bf16* __restrict__ TOWERIN, int B)
{
  int b = blockIdx.x * 8 + (threadIdx.x >> 5);
  int lane = threadIdx.x & 31;
  if (b >= B) return;
  const float* q = tq + (size_t)task[b] * 400;
  float qv[13];
#pragma unroll
  for (int i = 0; i < 13; ++i) { int d = lane + 32 * i; qv[i] = (d < 400) ? q[d] : 0.0f; }
  float sc[NEXP];
#pragma unroll
  for (int n = 0; n < NEXP; ++n) {
    const bf16* kp = KEY + ((size_t)n * B + b) * DPAD;
    float s = 0.0f;
#pragma unroll
    for (int i = 0; i < 13; ++i) { int d = lane + 32 * i; if (d < 400) s += (float)kp[d] * qv[i]; }
#pragma unroll
    for (int off = 16; off >= 1; off >>= 1) s += __shfl_xor(s, off, 32);
    sc[n] = s;                                 // butterfly: same value in all lanes
  }
  float mx = sc[0];
#pragma unroll
  for (int n = 1; n < NEXP; ++n) mx = fmaxf(mx, sc[n]);
  float sum = 0.0f;
#pragma unroll
  for (int n = 0; n < NEXP; ++n) { sc[n] = __expf(sc[n] - mx); sum += sc[n]; }
  float inv = 1.0f / sum;
#pragma unroll
  for (int n = 0; n < NEXP; ++n) sc[n] *= inv;
#pragma unroll
  for (int i = 0; i < 13; ++i) {
    int d = lane + 32 * i;
    float a = 0.0f;
    if (d < 400) {
#pragma unroll
      for (int n = 0; n < NEXP; ++n)
        a += sc[n] * (float)VAL[((size_t)n * B + b) * DPAD + d];
    }
    TOWERIN[(size_t)b * DPAD + d] = (bf16)a;   // pads (d>=400) written as 0
  }
}

// ---- final D->1 layer (wave32 per row, f32 out) -----------------------------
__global__ __launch_bounds__(256) void tower_out(
    const bf16* __restrict__ T2, const float* __restrict__ w3,
    const float* __restrict__ b3, float* __restrict__ out, int B)
{
  int b = blockIdx.x * 8 + (threadIdx.x >> 5);
  int lane = threadIdx.x & 31;
  if (b >= B) return;
  float s = 0.0f;
  for (int d = lane; d < 400; d += 32)
    s += (float)T2[(size_t)b * DPAD + d] * w3[d];
#pragma unroll
  for (int off = 16; off >= 1; off >>= 1) s += __shfl_xor(s, off, 32);
  if (lane == 0) out[b] = s + b3[0];
}

// ---------------------------------------------------------------------------
extern "C" void kernel_launch(void* const* d_in, const int* in_sizes, int n_in,
                              void* d_out, int out_size, void* d_ws, size_t ws_size,
                              hipStream_t stream) {
  const int B = BATCH;
  const float* obs    = (const float*)d_in[0];
  const float* act    = (const float*)d_in[1];
  const int*   task   = (const int*)  d_in[2];
  const float* bb_w1  = (const float*)d_in[3];
  const float* bb_b1  = (const float*)d_in[4];
  const float* bb_w2  = (const float*)d_in[5];
  const float* bb_b2  = (const float*)d_in[6];
  const float* ex_w1  = (const float*)d_in[7];
  const float* ex_b1  = (const float*)d_in[8];
  const float* ex_w2  = (const float*)d_in[9];
  const float* ex_b2  = (const float*)d_in[10];
  const float* key_m  = (const float*)d_in[11];
  const float* val_m  = (const float*)d_in[12];
  const float* tq     = (const float*)d_in[13];
  const float* tw_w1  = (const float*)d_in[14];
  const float* tw_b1  = (const float*)d_in[15];
  const float* tw_w2  = (const float*)d_in[16];
  const float* tw_b2  = (const float*)d_in[17];
  const float* tw_w3  = (const float*)d_in[18];
  const float* tw_b3  = (const float*)d_in[19];
  float* out = (float*)d_out;

  // ---- workspace carve-up (256B aligned) ----
  size_t off = 0;
  auto carve = [&](size_t bytes) {
    void* p = (char*)d_ws + off;
    off += (bytes + 255) & ~(size_t)255;
    return p;
  };
  const size_t SBe = (size_t)B * DPAD;          // activation elements per [B,416]
  const size_t SBb = SBe * sizeof(bf16);
  const size_t WEb = (size_t)NV * DPAD * sizeof(bf16);

  bf16* XA    = (bf16*)carve((size_t)B * 64 * sizeof(bf16));
  bf16* w_bb1 = (bf16*)carve((size_t)NV * 64 * sizeof(bf16));
  bf16* w_bb2 = (bf16*)carve(WEb);
  bf16* w_ex1 = (bf16*)carve(WEb * NEXP);
  bf16* w_ex2 = (bf16*)carve(WEb * NEXP);
  bf16* w_key = (bf16*)carve(WEb * NEXP);
  bf16* w_val = (bf16*)carve(WEb * NEXP);
  bf16* w_tw1 = (bf16*)carve(WEb);
  bf16* w_tw2 = (bf16*)carve(WEb);
  bf16* H1    = (bf16*)carve(SBb);              // later: TOWERIN
  bf16* H     = (bf16*)carve(SBb);              // later: T1
  bf16* E1    = (bf16*)carve(SBb * NEXP);       // later: KEY
  bf16* E     = (bf16*)carve(SBb * NEXP);       // later: T2
  bf16* VALb  = (bf16*)carve(SBb * NEXP);
  (void)ws_size; (void)in_sizes; (void)n_in; (void)out_size;

  const long wB = (long)NV * DPAD;              // weight batch stride (elements)
  const long sB = (long)NV * NV;                // fp32 source batch stride

  // ---- weight/input conversion ----
  build_x<<<(B * 64 + 255) / 256, 256, 0, stream>>>(obs, act, XA, B);
  {
    int t1 = NV * 64;
    convert_wt<<<(t1 + 255) / 256, 256, 0, stream>>>(bb_w1, w_bb1, 43, NV, 64, 0, 0, 0, t1);
    int t = NV * DPAD;
    convert_wt<<<(t + 255) / 256, 256, 0, stream>>>(bb_w2, w_bb2, NV, NV, DPAD, 0, 0, 0, t);
    convert_wt<<<(t + 255) / 256, 256, 0, stream>>>(tw_w1, w_tw1, NV, NV, DPAD, 0, 0, 0, t);
    convert_wt<<<(t + 255) / 256, 256, 0, stream>>>(tw_w2, w_tw2, NV, NV, DPAD, 0, 0, 0, t);
    int t8 = NEXP * NV * DPAD;
    convert_wt<<<(t8 + 255) / 256, 256, 0, stream>>>(ex_w1, w_ex1, NV, NV, DPAD, 0, sB, wB, t8);
    convert_wt<<<(t8 + 255) / 256, 256, 0, stream>>>(ex_w2, w_ex2, NV, NV, DPAD, 0, sB, wB, t8);
    convert_wt<<<(t8 + 255) / 256, 256, 0, stream>>>(key_m, w_key, NV, NV, DPAD, 1, sB, wB, t8);
    convert_wt<<<(t8 + 255) / 256, 256, 0, stream>>>(val_m, w_val, NV, NV, DPAD, 1, sB, wB, t8);
  }

  const dim3 blk(256);
  const dim3 g1(B / BM, (NV + BN - 1) / BN, 1);
  const dim3 g8(B / BM, (NV + BN - 1) / BN, NEXP);

  // backbone
  gemm_bf16_wmma<<<g1, blk, 0, stream>>>(XA, w_bb1, bb_b1, H1,
      NV, 64, 64, DPAD, 0, 0, 0, 0, 1);
  gemm_bf16_wmma<<<g1, blk, 0, stream>>>(H1, w_bb2, bb_b2, H,
      NV, DPAD, DPAD, DPAD, 0, 0, 0, 0, 1);
  // experts (batched over grid.z = 8)
  gemm_bf16_wmma<<<g8, blk, 0, stream>>>(H, w_ex1, ex_b1, E1,
      NV, DPAD, DPAD, DPAD, 0, wB, NV, (long)SBe, 1);
  gemm_bf16_wmma<<<g8, blk, 0, stream>>>(E1, w_ex2, ex_b2, E,
      NV, DPAD, DPAD, DPAD, (long)SBe, wB, NV, (long)SBe, 1);
  // key/value projections (no bias, no relu); KEY reuses E1's buffer
  bf16* KEY = E1;
  gemm_bf16_wmma<<<g8, blk, 0, stream>>>(E, w_key, nullptr, KEY,
      NV, DPAD, DPAD, DPAD, (long)SBe, wB, 0, (long)SBe, 0);
  gemm_bf16_wmma<<<g8, blk, 0, stream>>>(E, w_val, nullptr, VALb,
      NV, DPAD, DPAD, DPAD, (long)SBe, wB, 0, (long)SBe, 0);
  // attention combine -> TOWERIN (reuses H1)
  bf16* TOWERIN = H1;
  attn_combine<<<(B + 7) / 8, blk, 0, stream>>>(KEY, VALb, task, tq, TOWERIN, B);
  // tower
  bf16* T1 = H;
  gemm_bf16_wmma<<<g1, blk, 0, stream>>>(TOWERIN, w_tw1, tw_b1, T1,
      NV, DPAD, DPAD, DPAD, 0, 0, 0, 0, 1);
  bf16* T2 = E;
  gemm_bf16_wmma<<<g1, blk, 0, stream>>>(T1, w_tw2, tw_b2, T2,
      NV, DPAD, DPAD, DPAD, 0, 0, 0, 0, 1);
  tower_out<<<(B + 7) / 8, blk, 0, stream>>>(T2, tw_w3, tw_b3, out, B);
}
